// MinkUNetBase_3590592660321
// MI455X (gfx1250) — matrix operand order repaired and checked by reading, not compiled, not run on hardware
//
#include <hip/hip_runtime.h>
#include <hip/hip_bf16.h>

#define VN 8192
#define VC 256
#define VH 8
#define VHD 32
#define VEPS 1e-6f
#define CK 288                      // padded reduction dim for QKV proj (9*32)
#define NQKV 768                    // Q|K|V concatenated output columns
#define LOG2E 1.4426950408889634f
#define CLIP2 144.26950408889634f   // 100 * log2(e)

typedef __attribute__((ext_vector_type(16))) __bf16 v16bf;
typedef __attribute__((ext_vector_type(8)))  float  v8f;

struct __attribute__((aligned(16))) bfx8 { __bf16 h[8]; };
struct __attribute__((aligned(8)))  bfx4 { __bf16 h[4]; };

static __device__ __forceinline__ v16bf ld16(const __bf16* p) {
    bfx8 a = *(const bfx8*)p;
    bfx8 b = *(const bfx8*)(p + 8);
    v16bf r;
#pragma unroll
    for (int e = 0; e < 8; e++) { r[e] = a.h[e]; r[8 + e] = b.h[e]; }
    return r;
}
static __device__ __forceinline__ v16bf ld8x2(const __bf16* p0, const __bf16* p1) {
    bfx8 a = *(const bfx8*)p0;
    bfx8 b = *(const bfx8*)p1;
    v16bf r;
#pragma unroll
    for (int e = 0; e < 8; e++) { r[e] = a.h[e]; r[8 + e] = b.h[e]; }
    return r;
}
static __device__ __forceinline__ float clip2(float x) {
#if __has_builtin(__builtin_amdgcn_fmed3f)
    return __builtin_amdgcn_fmed3f(x, -CLIP2, CLIP2);   // single v_med3, no canon
#else
    return fminf(fmaxf(x, -CLIP2), CLIP2);
#endif
}
#if __has_builtin(__builtin_amdgcn_exp2f)
#define EXP2F(x) __builtin_amdgcn_exp2f(x)
#else
#define EXP2F(x) exp2f(x)
#endif
// key permutation inside a 64-key block: score j of lane lg sits at position
// 4*lg + j, so P rows store as one b64 per lane; Vt is laid out to match.
static __device__ __forceinline__ int vperm64(int c) {
    return ((c >> 4) & 1) | (((c >> 5) & 1) << 1) | ((c & 15) << 2);
}

// ---------------------------------------------------------------------------
// Prep 1: L2-normalize feats, append clipped coords, zero-pad to CK, bf16.
// ---------------------------------------------------------------------------
__global__ __launch_bounds__(256) void mink_prep_xc(
    const float* __restrict__ feats, const int* __restrict__ coords,
    __bf16* __restrict__ xcb)
{
    const int n = blockIdx.x;
    const int t = threadIdx.x;
    __shared__ float part[8];

    float f  = feats[(size_t)n * VC + t];
    float ss = f * f;
#pragma unroll
    for (int m = 1; m < 32; m <<= 1) ss += __shfl_xor(ss, m, 32);
    if ((t & 31) == 0) part[t >> 5] = ss;
    __syncthreads();
    float tot = 0.f;
#pragma unroll
    for (int i = 0; i < 8; i++) tot += part[i];
    const float inv = 1.0f / (sqrtf(tot) + VEPS);

    xcb[(size_t)n * CK + t] = (__bf16)(f * inv);
    if (t < CK - VC) {                       // 32 pad columns
        float v = 0.f;
        if (t < 3) {
            float cf = (float)coords[(size_t)n * 4 + 1 + t];
            v = fminf(fmaxf(cf, -100.f), 100.f);
        }
        xcb[(size_t)n * CK + VC + t] = (__bf16)v;
    }
}

// ---------------------------------------------------------------------------
// Prep 2: pack Wq|Wk|Wv transposed + zero-padded:  Wcatb[col][k], bf16.
// ---------------------------------------------------------------------------
__global__ __launch_bounds__(288) void mink_pack_w(
    const float* __restrict__ Wq, const float* __restrict__ Wk,
    const float* __restrict__ Wv, __bf16* __restrict__ Wcatb)
{
    const int j = blockIdx.x;          // output column 0..767
    const int k = threadIdx.x;         // reduction index 0..287
    const int mat = j >> 8;
    const int jj  = j & 255;
    const int h = jj >> 5, d = jj & 31;
    const float* W = (mat == 0) ? Wq : (mat == 1) ? Wk : Wv;
    float v = (k < VC + 3) ? W[((size_t)h * (VC + 3) + k) * VHD + d] : 0.f;
    Wcatb[(size_t)j * CK + k] = (__bf16)v;
}

// ---------------------------------------------------------------------------
// Prep 3: transpose + bf16-convert Wout -> WoutB[col][k].
// ---------------------------------------------------------------------------
__global__ __launch_bounds__(256) void mink_wout_t(
    const float* __restrict__ Wout, __bf16* __restrict__ WoutB)
{
    const int col = blockIdx.x;
    const int k   = threadIdx.x;
    WoutB[(size_t)col * VC + k] = (__bf16)Wout[(size_t)k * VC + col];
}

// ---------------------------------------------------------------------------
// Kernel 1: QKV projection as a WMMA GEMM [VN x CK] x [CK x NQKV].
// Epilogue scatters into Qb (pre-scaled by log2e/sqrt(HD)), Kb, permuted Vt.
// ---------------------------------------------------------------------------
__global__ __launch_bounds__(256) void mink_qkv_wmma(
    const __bf16* __restrict__ xcb, const __bf16* __restrict__ Wcatb,
    const float* __restrict__ bq, const float* __restrict__ bk,
    const float* __restrict__ bv,
    __bf16* __restrict__ Qb, __bf16* __restrict__ Kb, __bf16* __restrict__ Vt)
{
    const int lane = threadIdx.x & 31;
    const int wid  = threadIdx.x >> 5;
    const int lg   = lane & 15;
    const int hi   = lane >> 4;
    const int o    = hi ? 8 : 0;
    const int doff = hi ? 16 : 0;

    const int tile    = blockIdx.x * 8 + wid;       // 512*48 tiles
    const int rowbase = (tile / 48) * 16;
    const int colbase = (tile % 48) * 16;

    v8f acc = {};
#pragma unroll
    for (int kc = 0; kc < CK; kc += 32) {
        const __bf16* ar = xcb   + (size_t)(rowbase + lg) * CK + kc;
        const __bf16* br = Wcatb + (size_t)(colbase + lg) * CK + kc + doff;
        acc = __builtin_amdgcn_wmma_f32_16x16x32_bf16(false, ld16(ar + doff),
                                                      false, ld16(br), (short)0, acc, false, false);
    }

    const int mat = colbase >> 8;                    // 0=Q 1=K 2=V (uniform per wave)
    const int jj  = (colbase & 255) | lg;
    const int h = jj >> 5, d = jj & 31;
    const float* bb = (mat == 0) ? bq : (mat == 1) ? bk : bv;
    const float bias = bb[jj];
    const float qsc  = 0.17677669529663687f * LOG2E; // 1/sqrt(32) * log2(e)

#pragma unroll
    for (int r = 0; r < 8; r++) {
        const int row = rowbase + r + 8 * hi;
        const float val = acc[r] + bias;
        if (mat == 0) {
            Qb[((size_t)h * VN + row) * VHD + d] = (__bf16)(val * qsc);
        } else if (mat == 1) {
            Kb[((size_t)h * VN + row) * VHD + d] = (__bf16)val;
        } else {
            const int pos = (row & ~63) | vperm64(row & 63);
            Vt[((size_t)h * VHD + d) * VN + pos] = (__bf16)val;
        }
    }
}

// ---------------------------------------------------------------------------
// Flash-attention step over one 64-key LDS tile (8 WMMAs + log2-domain
// online softmax; P re-staged through a wave-private LDS tile as b64 packs).
// ---------------------------------------------------------------------------
static __device__ __forceinline__ void flash_step(
    const __bf16 (* __restrict__ kb)[VHD], const __bf16 (* __restrict__ vb)[64],
    __bf16 (* __restrict__ pb)[64],
    const v16bf aQ, v8f& acc0, v8f& acc1, float* m, float* lp,
    const int lg, const int hi, const int o, const int doff)
{
    // V B-tiles first: DS latency hides under the softmax VALU work
    v16bf bV0a = ld16(&vb[lg]     [doff]);
    v16bf bV0b = ld16(&vb[16 + lg][doff]);
    v16bf bV1a = ld16(&vb[lg]     [32 + doff]);
    v16bf bV1b = ld16(&vb[16 + lg][32 + doff]);

    v8f s[4];
#pragma unroll
    for (int j = 0; j < 4; j++) {
        v16bf bK = ld16(&kb[j * 16 + lg][doff]);
        v8f z = {};
        s[j] = __builtin_amdgcn_wmma_f32_16x16x32_bf16(false, aQ, false, bK, (short)0, z, false, false);
    }

#pragma unroll
    for (int r = 0; r < 8; r++) {
        float a0 = clip2(s[0][r]);
        float a1 = clip2(s[1][r]);
        float a2 = clip2(s[2][r]);
        float a3 = clip2(s[3][r]);
        float tmax = fmaxf(fmaxf(a0, a1), fmaxf(a2, a3));
#pragma unroll
        for (int mm = 1; mm < 16; mm <<= 1) tmax = fmaxf(tmax, __shfl_xor(tmax, mm, 32));
        const float mn = fmaxf(m[r], tmax);
        const float sc = EXP2F(m[r] - mn);
        m[r] = mn;
        float p0 = EXP2F(a0 - mn);
        float p1 = EXP2F(a1 - mn);
        float p2 = EXP2F(a2 - mn);
        float p3 = EXP2F(a3 - mn);
        lp[r] = lp[r] * sc + ((p0 + p1) + (p2 + p3));
        acc0[r] *= sc;
        acc1[r] *= sc;
        bfx4 pk;
        pk.h[0] = (__bf16)p0; pk.h[1] = (__bf16)p1;
        pk.h[2] = (__bf16)p2; pk.h[3] = (__bf16)p3;
        *(bfx4*)&pb[r + 8 * hi][4 * lg] = pk;        // one ds_store_b64 per row
    }

    const __bf16* pr = &pb[lg][0];
    v16bf aP0 = ld16(pr + doff);                     // positions  0..31
    v16bf aP1 = ld16(pr + 32 + doff);                // positions 32..63

    acc0 = __builtin_amdgcn_wmma_f32_16x16x32_bf16(false, aP0, false, bV0a, (short)0, acc0, false, false);
    acc1 = __builtin_amdgcn_wmma_f32_16x16x32_bf16(false, aP0, false, bV0b, (short)0, acc1, false, false);
    acc0 = __builtin_amdgcn_wmma_f32_16x16x32_bf16(false, aP1, false, bV1a, (short)0, acc0, false, false);
    acc1 = __builtin_amdgcn_wmma_f32_16x16x32_bf16(false, aP1, false, bV1b, (short)0, acc1, false, false);
}

// ---------------------------------------------------------------------------
// Kernel 2: flash attention.  8 waves share one head; 64-key double-buffered
// LDS tiles; last iteration peeled so staging is branch-free.
// ---------------------------------------------------------------------------
__global__ __launch_bounds__(256) void mink_flash_attn(
    const __bf16* __restrict__ Qb, const __bf16* __restrict__ Kb,
    const __bf16* __restrict__ Vt, __bf16* __restrict__ catbuf)
{
    const int t    = threadIdx.x;
    const int lane = t & 31;
    const int wid  = t >> 5;
    const int lg   = lane & 15;
    const int hi   = lane >> 4;
    const int o    = hi ? 8 : 0;
    const int doff = hi ? 16 : 0;

    const int blocksPerHead = (VN / 16) / 8;            // 64
    const int h     = blockIdx.x / blocksPerHead;
    const int qbase = ((blockIdx.x % blocksPerHead) * 8 + wid) * 16;

    __shared__ __attribute__((aligned(16))) __bf16 kbuf[2][64][VHD];  // 8 KB
    __shared__ __attribute__((aligned(16))) __bf16 vbuf[2][VHD][64];  // 8 KB
    __shared__ __attribute__((aligned(16))) __bf16 pbuf[8][16][64];   // 16 KB

    const __bf16* __restrict__ Kh = Kb + (size_t)h * VN * VHD;
    const __bf16* __restrict__ Vh = Vt + (size_t)h * VHD * VN;

    const int krow = t >> 2, kchk = (t & 3) * 8;
    const int vrow = t >> 3, vchk = (t & 7) * 8;

    v16bf aQ;
    {
        const __bf16* qrow = Qb + ((size_t)h * VN + (qbase + lg)) * VHD;
        aQ = ld16(qrow + doff);
    }

    float m[8], lp[8];
    v8f acc0 = {}; v8f acc1 = {};
#pragma unroll
    for (int r = 0; r < 8; r++) { m[r] = -3.0e38f; lp[r] = 0.f; }

    *(bfx8*)&kbuf[0][krow][kchk] = *(const bfx8*)(Kh + (size_t)krow * VHD + kchk);
    *(bfx8*)&vbuf[0][vrow][vchk] = *(const bfx8*)(Vh + (size_t)vrow * VN + vchk);
    __syncthreads();

    for (int kb = 0; kb < VN - 64; kb += 64) {
        const int buf = (kb >> 6) & 1;
        // issue next-tile global loads now; the loadcnt wait lands at the LDS
        // stores after the compute, so L2 latency overlaps the 8 WMMAs.
        bfx8 nk = *(const bfx8*)(Kh + (size_t)(kb + 64 + krow) * VHD + kchk);
        bfx8 nv = *(const bfx8*)(Vh + (size_t)vrow * VN + (kb + 64) + vchk);

        flash_step(kbuf[buf], vbuf[buf], pbuf[wid], aQ, acc0, acc1, m, lp, lg, hi, o, doff);

        *(bfx8*)&kbuf[buf ^ 1][krow][kchk] = nk;
        *(bfx8*)&vbuf[buf ^ 1][vrow][vchk] = nv;
        __syncthreads();
    }
    flash_step(kbuf[1], vbuf[1], pbuf[wid], aQ, acc0, acc1, m, lp, lg, hi, o, doff);

#pragma unroll
    for (int r = 0; r < 8; r++) {
        float lr = lp[r];
#pragma unroll
        for (int mm = 1; mm < 16; mm <<= 1) lr += __shfl_xor(lr, mm, 32);
        const float invl = 1.0f / (lr + VEPS);
        const int row = qbase + r + 8 * hi;
        __bf16* orow = catbuf + (size_t)row * VC + h * VHD;
        orow[lg]      = (__bf16)(acc0[r] * invl);
        orow[16 + lg] = (__bf16)(acc1[r] * invl);
    }
}

// ---------------------------------------------------------------------------
// Kernel 3: out = cat @ Wout + bout + feats via WMMA (16x16 tile per wave).
// ---------------------------------------------------------------------------
__global__ __launch_bounds__(256) void mink_out_wmma(
    const __bf16* __restrict__ catb, const __bf16* __restrict__ WoutB,
    const float* __restrict__ bout, const float* __restrict__ feats,
    float* __restrict__ out)
{
    const int lane = threadIdx.x & 31;
    const int wid  = threadIdx.x >> 5;
    const int lg   = lane & 15;
    const int hi   = lane >> 4;
    const int o    = hi ? 8 : 0;

    const int tile    = blockIdx.x * 8 + wid;      // 512*16 tiles
    const int rowbase = (tile >> 4) << 4;
    const int colbase = (tile & 15) << 4;

    v8f acc = {};
#pragma unroll
    for (int kc = 0; kc < VC; kc += 32) {
        const __bf16* ar = catb  + (size_t)(rowbase + lg) * VC + kc;
        const __bf16* br = WoutB + (size_t)(colbase + lg) * VC + kc + (hi ? 16 : 0);
        acc = __builtin_amdgcn_wmma_f32_16x16x32_bf16(false, ld16(ar + (hi ? 16 : 0)),
                                                      false, ld16(br), (short)0, acc, false, false);
    }
#pragma unroll
    for (int r = 0; r < 8; r++) {
        const int row = rowbase + r + 8 * hi;
        const int col = colbase + lg;
        out[(size_t)row * VC + col] = acc[r] + bout[col] + feats[(size_t)row * VC + col];
    }
}

// ---------------------------------------------------------------------------
extern "C" void kernel_launch(void* const* d_in, const int* in_sizes, int n_in,
                              void* d_out, int out_size, void* d_ws, size_t ws_size,
                              hipStream_t stream) {
    const float* feats  = (const float*)d_in[0];
    const int*   coords = (const int*)  d_in[1];
    const float* Wq     = (const float*)d_in[2];
    const float* bq     = (const float*)d_in[3];
    const float* Wk     = (const float*)d_in[4];
    const float* bk     = (const float*)d_in[5];
    const float* Wv     = (const float*)d_in[6];
    const float* bv     = (const float*)d_in[7];
    const float* Wout   = (const float*)d_in[8];
    const float* bout   = (const float*)d_in[9];
    float*       out    = (float*)d_out;

    char* ws = (char*)d_ws;
    __bf16* xcb   = (__bf16*)(ws);                          // 4.5 MB
    __bf16* Qb    = (__bf16*)(ws + ((size_t)5  << 20));     // 4 MB
    __bf16* Kb    = (__bf16*)(ws + ((size_t)9  << 20));     // 4 MB
    __bf16* Vt    = (__bf16*)(ws + ((size_t)13 << 20));     // 4 MB (key-permuted)
    __bf16* catb  = (__bf16*)(ws + ((size_t)17 << 20));     // 4 MB
    __bf16* WoutB = (__bf16*)(ws + ((size_t)21 << 20));     // 128 KB
    __bf16* Wcatb = (__bf16*)(ws + ((size_t)22 << 20));     // 432 KB

    mink_prep_xc<<<VN, 256, 0, stream>>>(feats, coords, xcb);
    mink_pack_w<<<NQKV, 288, 0, stream>>>(Wq, Wk, Wv, Wcatb);
    mink_wout_t<<<VC, 256, 0, stream>>>(Wout, WoutB);

    const int qkvBlocks = (VN / 16) * (NQKV / 16) / 8;      // 3072
    mink_qkv_wmma<<<qkvBlocks, 256, 0, stream>>>(xcb, Wcatb, bq, bk, bv, Qb, Kb, Vt);

    const int attnBlocks = VH * ((VN / 16) / 8);            // 512
    mink_flash_attn<<<attnBlocks, 256, 0, stream>>>(Qb, Kb, Vt, catb);

    const int outBlocks = (VN / 16) * (VC / 16) / 8;        // 1024
    mink_out_wmma<<<outBlocks, 256, 0, stream>>>(catb, WoutB, bout, feats, out);
}
